// AtomAttentionPairBias_72327249264737
// MI455X (gfx1250) — matrix operand order repaired and checked by reading, not compiled, not run on hardware
//
#include <hip/hip_runtime.h>
#include <hip/hip_bf16.h>
#include <cstdint>

typedef __attribute__((ext_vector_type(16))) _Float16 v16h;
typedef __attribute__((ext_vector_type(8)))  float    v8f;

#define NSEQ 2048
#define CIN  128

union AFrag { v16h v; uint32_t u[8]; };

__device__ __forceinline__ v8f wmma16(const AFrag& a, const AFrag& b, v8f c) {
  return __builtin_amdgcn_wmma_f32_16x16x32_f16(false, a.v, false, b.v, (short)0, c, false, false);
}

// A fragment (16x32 f16, M x K). rowbase = &A[row(lane&15)][k0]; hiHalf = lane>=16.
// lanes 0-15: K {0..15}; lanes 16-31: K {16..31} (pairs packed per u32).
__device__ __forceinline__ void load_afrag(AFrag& f, const _Float16* rowbase, int hiHalf) {
  const uint32_t* ar = (const uint32_t*)rowbase;
  int o = hiHalf * 8;
#pragma unroll
  for (int p = 0; p < 8; ++p) { f.u[p] = ar[o + p]; }
}

// B fragment (32x16 f16, K x N) from Bt stored [N][K].
// colbase = &Bt[n(lane&15)][k0 + (hiHalf?16:0)]; 16 consecutive K halves.
__device__ __forceinline__ void load_bfrag(AFrag& f, const _Float16* colbase) {
  const uint32_t* br = (const uint32_t*)colbase;
#pragma unroll
  for (int p = 0; p < 8; ++p) f.u[p] = br[p];
}

__device__ __forceinline__ v8f vzero8() {
  v8f z = {0.f, 0.f, 0.f, 0.f, 0.f, 0.f, 0.f, 0.f};
  return z;
}

// ---------------- weight prep: Wt[n][k] = W[k][n] * (scale?scale[k]:1), f32 -> f16 ----------------
__global__ __launch_bounds__(256) void wprep_kernel(const float* __restrict__ W,
                                                    const float* __restrict__ scale,
                                                    _Float16* __restrict__ Wt) {
  int i = blockIdx.x * 256 + threadIdx.x;
  if (i >= CIN * CIN) return;
  int n = i >> 7, k = i & 127;
  float s = scale ? scale[k] : 1.0f;
  Wt[n * CIN + k] = (_Float16)(W[k * CIN + n] * s);
}

// ---------------- row LayerNorms: xn (f32), sn = LN(cond) (f16), sc = raw cond (f16) -------------
__device__ __forceinline__ float block_sum128(float v, float* red, int t) {
  red[t] = v; __syncthreads();
#pragma unroll
  for (int s = 64; s > 0; s >>= 1) { if (t < s) red[t] += red[t + s]; __syncthreads(); }
  float r = red[0]; __syncthreads();
  return r;
}

__global__ __launch_bounds__(128) void ln_kernel(const float* __restrict__ act,
                                                 const float* __restrict__ cond,
                                                 float* __restrict__ xn,
                                                 _Float16* __restrict__ sn,
                                                 _Float16* __restrict__ sc) {
  __shared__ float red[128];
  int row = blockIdx.x, t = threadIdx.x;
  float a = act[row * CIN + t];
  float c = cond[row * CIN + t];

  float ma = block_sum128(a, red, t) * (1.0f / 128.0f);
  float da = a - ma;
  float va = block_sum128(da * da, red, t) * (1.0f / 128.0f);
  xn[row * CIN + t] = da * rsqrtf(va + 1e-5f);

  float mc = block_sum128(c, red, t) * (1.0f / 128.0f);
  float dc = c - mc;
  float vc = block_sum128(dc * dc, red, t) * (1.0f / 128.0f);
  sn[row * CIN + t] = (_Float16)(dc * rsqrtf(vc + 1e-5f));
  sc[row * CIN + t] = (_Float16)c;
}

// ---------------- fused AdaLN: a = sigmoid(sn@Wg + bg) * xn + sn@Ws  (lns folded in Wt) ----------
__global__ __launch_bounds__(256) void adaln_gemm_kernel(const _Float16* __restrict__ A,
                                                         const _Float16* __restrict__ WtG,
                                                         const float* __restrict__ bg,
                                                         const _Float16* __restrict__ WtS,
                                                         const float* __restrict__ xn,
                                                         _Float16* __restrict__ out) {
  int wave = threadIdx.x >> 5, lane = threadIdx.x & 31;
  int tile = blockIdx.x * 8 + wave;
  int mt = tile >> 3, nt = tile & 7;
  int mlane = lane & 15, hiHalf = (lane >> 4) & 1;

  v8f accG = vzero8(), accS = vzero8();
  const _Float16* Arow = A + (mt * 16 + mlane) * CIN;
  const _Float16* BrowG = WtG + (nt * 16 + mlane) * CIN + hiHalf * 16;
  const _Float16* BrowS = WtS + (nt * 16 + mlane) * CIN + hiHalf * 16;
#pragma unroll
  for (int ks = 0; ks < 4; ++ks) {
    AFrag af, bg_, bs_;
    load_afrag(af, Arow + ks * 32, hiHalf);
    load_bfrag(bg_, BrowG + ks * 32);
    load_bfrag(bs_, BrowS + ks * 32);
    accG = wmma16(af, bg_, accG);
    accS = wmma16(af, bs_, accS);
  }
  int n = nt * 16 + mlane;
  float bias = bg[n];
#pragma unroll
  for (int r = 0; r < 8; ++r) {
    int i = mt * 16 + hiHalf * 8 + r;
    float gate = 1.0f / (1.0f + __expf(-(accG[r] + bias)));
    out[i * CIN + n] = (_Float16)(gate * xn[i * CIN + n] + accS[r]);
  }
}

// ---------------- generic 2048x128 @ 128x128 GEMM with epilogue -----------------------------------
// mode 0: f32 out = acc(+bias); 1: f32 sigmoid(acc+bias); 2: f16 out; 3: f32 aux*(acc+bias)
__global__ __launch_bounds__(256) void gemm_epi_kernel(const _Float16* __restrict__ A,
                                                       const _Float16* __restrict__ Bt,
                                                       const float* __restrict__ bias,
                                                       const float* __restrict__ aux,
                                                       void* __restrict__ out, int mode) {
  int wave = threadIdx.x >> 5, lane = threadIdx.x & 31;
  int tile = blockIdx.x * 8 + wave;
  int mt = tile >> 3, nt = tile & 7;
  int mlane = lane & 15, hiHalf = (lane >> 4) & 1;

  v8f acc = vzero8();
  const _Float16* Arow = A + (mt * 16 + mlane) * CIN;
  const _Float16* Brow = Bt + (nt * 16 + mlane) * CIN + hiHalf * 16;
#pragma unroll
  for (int ks = 0; ks < 4; ++ks) {
    AFrag af, bf;
    load_afrag(af, Arow + ks * 32, hiHalf);
    load_bfrag(bf, Brow + ks * 32);
    acc = wmma16(af, bf, acc);
  }
  int n = nt * 16 + mlane;
  float bv = bias ? bias[n] : 0.0f;
#pragma unroll
  for (int r = 0; r < 8; ++r) {
    int i = mt * 16 + hiHalf * 8 + r;
    float v = acc[r] + bv;
    if (mode == 1) v = 1.0f / (1.0f + __expf(-v));
    else if (mode == 3) v = aux[i * CIN + n] * v;
    if (mode == 2) ((_Float16*)out)[i * CIN + n] = (_Float16)v;
    else ((float*)out)[i * CIN + n] = v;
  }
}

// ---------------- block attention: 1 WG per 32-query block, 8 waves = (4 heads x 2 qtiles) -------
// LDS carve:
//   Kraw [128][128] f16 32KB | Vt [128][128] f16 32KB | bias [32][128][4] f16 32KB |
//   S [8][16][128] f32 64KB (Vraw scratch overlays its first 32KB) | P [8][16][128] f16 32KB |
//   rsum[128] f32 | wbl[64] f32     => 197376 B dynamic LDS
#define ATTN_SMEM (128*128*2 + 128*128*2 + 32*128*4*2 + 8*16*128*4 + 8*16*128*2 + 128*4 + 64*4)

__global__ __launch_bounds__(256) void attn_kernel(const _Float16* __restrict__ qh,
                                                   const _Float16* __restrict__ kh,
                                                   const _Float16* __restrict__ vh,
                                                   const float* __restrict__ gproj,
                                                   const float* __restrict__ pair,
                                                   const float* __restrict__ lnz,
                                                   const float* __restrict__ Wb,
                                                   _Float16* __restrict__ oh) {
  extern __shared__ char smem[];
  _Float16* KrawL = (_Float16*)smem;                       // [jj][hc] row-major window of K
  _Float16* VtL   = KrawL + 128 * 128;                     // [h*32+c][jj] transposed V
  _Float16* biasL = VtL + 128 * 128;                       // [(q*128+jj)*4 + h]
  float*    SL    = (float*)(biasL + 32 * 128 * 4);        // wave-private [16][128]
  _Float16* VrawL = (_Float16*)SL;                         // scratch overlay: [jj][hc]
  _Float16* PL    = (_Float16*)(SL + 8 * 16 * 128);        // wave-private [16][128]
  float*    rsumL = (float*)(PL + 8 * 16 * 128);           // [wave*16+m]
  float*    wbl   = rsumL + 128;                           // lnz[c]*Wb[c][h]

  int t = threadIdx.x;
  int i0 = blockIdx.x * 32;
  int j0 = i0 - 48;

  if (t < 64) wbl[t] = lnz[t >> 2] * Wb[t];

  // ---- Phase A: async-stage K and V windows (row-major) into LDS; overlaps with bias compute ----
  {
    uint32_t ldsK = (uint32_t)(uintptr_t)KrawL;
    uint32_t ldsV = (uint32_t)(uintptr_t)VrawL;
#pragma unroll
    for (int cpy = 0; cpy < 8; ++cpy) {
      int idx = cpy * 256 + t;            // 2048 x b128 chunks per matrix
      int jj = idx >> 4;                  // window-local key row
      int cc = (idx & 15) * 16;           // byte offset within 256B row
      int j = j0 + jj;
      int jc = j < 0 ? 0 : (j >= NSEQ ? NSEQ - 1 : j);
      const char* gK = (const char*)(kh + (size_t)jc * CIN) + cc;
      const char* gV = (const char*)(vh + (size_t)jc * CIN) + cc;
      uint32_t lK = ldsK + (uint32_t)(jj * 256 + cc);
      uint32_t lV = ldsV + (uint32_t)(jj * 256 + cc);
      asm volatile("global_load_async_to_lds_b128 %0, %1, off"
                   :: "v"(lK), "v"(gK) : "memory");
      asm volatile("global_load_async_to_lds_b128 %0, %1, off"
                   :: "v"(lV), "v"(gV) : "memory");
    }
  }
  __syncthreads();   // wbl visible

  // ---- Phase B: pair bias LN(pair_act[i,j,:16]) . (lnz*Wb[:,h]) for all 4 heads (overlaps async)
  for (int idx = t; idx < 32 * 128; idx += 256) {
    int q = idx >> 7, jj = idx & 127;
    int i = i0 + q, j = j0 + jj;
    float b0 = 0.f, b1 = 0.f, b2 = 0.f, b3 = 0.f;
    if (j >= 0 && j < NSEQ) {
      const float* zr = pair + ((size_t)i * NSEQ + j) * 16;
      float x[16], m = 0.f;
#pragma unroll
      for (int c = 0; c < 16; ++c) { x[c] = zr[c]; m += x[c]; }
      m *= (1.0f / 16.0f);
      float vv = 0.f;
#pragma unroll
      for (int c = 0; c < 16; ++c) { float d = x[c] - m; vv += d * d; }
      float inv = rsqrtf(vv * (1.0f / 16.0f) + 1e-5f);
#pragma unroll
      for (int c = 0; c < 16; ++c) {
        float zn = (x[c] - m) * inv;
        b0 += zn * wbl[c * 4 + 0]; b1 += zn * wbl[c * 4 + 1];
        b2 += zn * wbl[c * 4 + 2]; b3 += zn * wbl[c * 4 + 3];
      }
    }
    biasL[idx * 4 + 0] = (_Float16)b0; biasL[idx * 4 + 1] = (_Float16)b1;
    biasL[idx * 4 + 2] = (_Float16)b2; biasL[idx * 4 + 3] = (_Float16)b3;
  }

  // ---- wait for this wave's async copies, then rendezvous so all K/V data is in LDS -------------
  asm volatile("s_wait_asynccnt 0x0" ::: "memory");
  __syncthreads();

  // ---- Phase C: LDS->LDS transpose of V into VtL (Vraw scratch freed afterwards) ----------------
  for (int idx = t; idx < 128 * 128; idx += 256) {
    int jj = idx >> 7, hc = idx & 127;
    VtL[hc * 128 + jj] = VrawL[jj * 128 + hc];
  }
  __syncthreads();   // Vraw scratch now reusable as SL

  int wave = t >> 5, lane = t & 31;
  int h = wave >> 1, qt = wave & 1;
  int qrow0 = i0 + qt * 16;
  int mlane = lane & 15, hiHalf = (lane >> 4) & 1;
  float* SW = SL + wave * 16 * 128;
  _Float16* PW = PL + wave * 16 * 128;

  // Q fragment: A 16x32 (queries x per-head channels), K=32=C in one step
  AFrag qa;
  load_afrag(qa, qh + (qrow0 + mlane) * CIN + h * 32, hiHalf);

  const float scale = 0.17677669529663687f; // 1/sqrt(32)
#pragma unroll
  for (int nt = 0; nt < 8; ++nt) {
    AFrag kb;
    load_bfrag(kb, KrawL + (nt * 16 + mlane) * CIN + h * 32 + hiHalf * 16);
    v8f acc = wmma16(qa, kb, vzero8());
#pragma unroll
    for (int r = 0; r < 8; ++r) {
      int m = hiHalf * 8 + r;
      int jj = nt * 16 + mlane;
      int j = j0 + jj;
      float s = acc[r] * scale + (float)biasL[((qt * 16 + m) * 128 + jj) * 4 + h];
      if (!(j >= 0 && j < NSEQ)) s = -1e9f;
      SW[m * 128 + jj] = s;
    }
  }
  __syncthreads();

  // per-wave softmax over 128-key window; keep unnormalized exp, fold 1/sum into O epilogue
  if (lane < 16) {
    int r = lane;
    float mx = -3.0e38f;
    for (int j = 0; j < 128; ++j) mx = fmaxf(mx, SW[r * 128 + j]);
    float sum = 0.f;
    for (int j = 0; j < 128; ++j) {
      float e = __expf(SW[r * 128 + j] - mx);
      sum += e;
      PW[r * 128 + j] = (_Float16)e;
    }
    rsumL[wave * 16 + r] = sum;
  }
  __syncthreads();

  // O = P @ V  (16q x 32c per wave, 2 N-tiles, 4 K-steps of 32 keys)
#pragma unroll
  for (int ct = 0; ct < 2; ++ct) {
    v8f acc = vzero8();
#pragma unroll
    for (int ks = 0; ks < 4; ++ks) {
      AFrag pa, vb;
      load_afrag(pa, PW + mlane * 128 + ks * 32, hiHalf);
      load_bfrag(vb, VtL + (h * 32 + ct * 16 + mlane) * 128 + ks * 32 + hiHalf * 16);
      acc = wmma16(pa, vb, acc);
    }
#pragma unroll
    for (int r = 0; r < 8; ++r) {
      int m = hiHalf * 8 + r;
      int i = qrow0 + m;
      int c = h * 32 + ct * 16 + mlane;
      float val = acc[r] / rsumL[wave * 16 + m];
      float gv = gproj[i * CIN + c];
      val *= 1.0f / (1.0f + __expf(-gv));
      oh[i * CIN + c] = (_Float16)val;
    }
  }
}

// ---------------------------------------------------------------------------------------------
extern "C" void kernel_launch(void* const* d_in, const int* in_sizes, int n_in,
                              void* d_out, int out_size, void* d_ws, size_t ws_size,
                              hipStream_t stream) {
  (void)in_sizes; (void)n_in; (void)out_size; (void)ws_size;
  const float* single_act  = (const float*)d_in[0];
  const float* pair_act    = (const float*)d_in[1];
  const float* single_cond = (const float*)d_in[2];
  // d_in[3] = block_mask (recomputed analytically; window is [blk-48, blk+80))
  const float* lns_q   = (const float*)d_in[4];
  const float* Wgate_q = (const float*)d_in[5];
  const float* bgate_q = (const float*)d_in[6];
  const float* Wskip_q = (const float*)d_in[7];
  const float* lns_k   = (const float*)d_in[8];
  const float* Wgate_k = (const float*)d_in[9];
  const float* bgate_k = (const float*)d_in[10];
  const float* Wskip_k = (const float*)d_in[11];
  const float* lnz_w   = (const float*)d_in[12];
  const float* Wq  = (const float*)d_in[13];
  const float* bq  = (const float*)d_in[14];
  const float* Wk  = (const float*)d_in[15];
  const float* Wv  = (const float*)d_in[16];
  const float* Wg  = (const float*)d_in[17];
  const float* Wb  = (const float*)d_in[18];
  const float* Wo  = (const float*)d_in[19];
  const float* Wgs = (const float*)d_in[20];
  const float* bgs = (const float*)d_in[21];

  // workspace bump allocator
  char* p = (char*)d_ws;
  auto alloc = [&](size_t bytes) { void* r = (void*)p; p += (bytes + 255) & ~size_t(255); return r; };
  const size_t WB = CIN * CIN * sizeof(_Float16);
  const size_t MB16 = (size_t)NSEQ * CIN * sizeof(_Float16);
  const size_t MB32 = (size_t)NSEQ * CIN * sizeof(float);

  _Float16* WtGq = (_Float16*)alloc(WB);
  _Float16* WtSq = (_Float16*)alloc(WB);
  _Float16* WtGk = (_Float16*)alloc(WB);
  _Float16* WtSk = (_Float16*)alloc(WB);
  _Float16* WtQ  = (_Float16*)alloc(WB);
  _Float16* WtK  = (_Float16*)alloc(WB);
  _Float16* WtV  = (_Float16*)alloc(WB);
  _Float16* WtG  = (_Float16*)alloc(WB);
  _Float16* WtO  = (_Float16*)alloc(WB);
  _Float16* WtGs = (_Float16*)alloc(WB);
  float*    xn   = (float*)alloc(MB32);
  _Float16* sn   = (_Float16*)alloc(MB16);
  _Float16* sc   = (_Float16*)alloc(MB16);
  _Float16* aq   = (_Float16*)alloc(MB16);
  _Float16* ak   = (_Float16*)alloc(MB16);
  _Float16* qhB  = (_Float16*)alloc(MB16);
  _Float16* khB  = (_Float16*)alloc(MB16);
  _Float16* vhB  = (_Float16*)alloc(MB16);
  float*    gbuf = (float*)alloc(MB32);
  float*    gateS= (float*)alloc(MB32);
  _Float16* ohB  = (_Float16*)alloc(MB16);

  // 1) weight prep (f32 -> f16, transposed to [N][K]; fold lns into gate/skip)
  wprep_kernel<<<64, 256, 0, stream>>>(Wgate_q, lns_q, WtGq);
  wprep_kernel<<<64, 256, 0, stream>>>(Wskip_q, lns_q, WtSq);
  wprep_kernel<<<64, 256, 0, stream>>>(Wgate_k, lns_k, WtGk);
  wprep_kernel<<<64, 256, 0, stream>>>(Wskip_k, lns_k, WtSk);
  wprep_kernel<<<64, 256, 0, stream>>>(Wq,  nullptr, WtQ);
  wprep_kernel<<<64, 256, 0, stream>>>(Wk,  nullptr, WtK);
  wprep_kernel<<<64, 256, 0, stream>>>(Wv,  nullptr, WtV);
  wprep_kernel<<<64, 256, 0, stream>>>(Wg,  nullptr, WtG);
  wprep_kernel<<<64, 256, 0, stream>>>(Wo,  nullptr, WtO);
  wprep_kernel<<<64, 256, 0, stream>>>(Wgs, nullptr, WtGs);

  // 2) row LayerNorms
  ln_kernel<<<NSEQ, 128, 0, stream>>>(single_act, single_cond, xn, sn, sc);

  // 3) fused AdaLN (gate + skip GEMMs share A fragments)
  adaln_gemm_kernel<<<128, 256, 0, stream>>>(sn, WtGq, bgate_q, WtSq, xn, aq);
  adaln_gemm_kernel<<<128, 256, 0, stream>>>(sn, WtGk, bgate_k, WtSk, xn, ak);

  // 4) projections
  gemm_epi_kernel<<<128, 256, 0, stream>>>(aq, WtQ, bq,      nullptr, (void*)qhB,  2);
  gemm_epi_kernel<<<128, 256, 0, stream>>>(ak, WtK, nullptr, nullptr, (void*)khB,  2);
  gemm_epi_kernel<<<128, 256, 0, stream>>>(ak, WtV, nullptr, nullptr, (void*)vhB,  2);
  gemm_epi_kernel<<<128, 256, 0, stream>>>(aq, WtG, nullptr, nullptr, (void*)gbuf, 0);
  gemm_epi_kernel<<<128, 256, 0, stream>>>(sc, WtGs, bgs,    nullptr, (void*)gateS,1);

  // 5) block attention (async K/V staging overlapped with pair-bias compute)
  attn_kernel<<<NSEQ / 32, 256, ATTN_SMEM, stream>>>(qhB, khB, vhB, gbuf,
                                                     pair_act, lnz_w, Wb, ohB);

  // 6) out = gateS * (o @ Wo)
  gemm_epi_kernel<<<128, 256, 0, stream>>>(ohB, WtO, nullptr, gateS, d_out, 3);
}
